// AttentionModule_32427003085576
// MI455X (gfx1250) — compile-verified, matrix operand order checked by reading, not run on hardware
//
#include <hip/hip_runtime.h>

// ---------------------------------------------------------------------------
// Cross-attention (B=4, T=L=2048, d_model=512, H=8, Dh=64) for gfx1250.
// All GEMMs on v_wmma_f32_16x16x32_f16; tile staging via
// global_load_async_to_lds_b128 (ASYNCcnt) with LDS double buffering.
// Weights pre-converted to f16; V stored (b,h,d,l)-transposed so every
// WMMA fragment is a contiguous ds_load_b128.
// ---------------------------------------------------------------------------

typedef __attribute__((ext_vector_type(16))) _Float16 v16h;
typedef __attribute__((ext_vector_type(4)))  _Float16 v4h;
typedef __attribute__((ext_vector_type(8)))  float    v8f;

#define D_MODEL 512
#define TSEQ    2048
#define LSEQ    2048
#define NHEADS  8
#define HDIM    64
#define BATCH   4

__device__ __forceinline__ v8f wmma16(v16h a, v16h b, v8f c) {
  return __builtin_amdgcn_wmma_f32_16x16x32_f16(
      false, a, false, b, (short)0, c, false, false);
}

// generic shared pointer -> 32-bit LDS byte offset (low 32 bits of aperture VA)
__device__ __forceinline__ unsigned lds_off(const void* p) {
  return (unsigned)(unsigned long long)p;
}

// one per-lane 16B async copy: global -> LDS (GV addressing), ASYNCcnt += 1/wave
__device__ __forceinline__ void async_b128(unsigned lds, const void* g) {
  asm volatile("global_load_async_to_lds_b128 %0, %1, off"
               :: "v"(lds), "v"((unsigned long long)g) : "memory");
}
#define WAIT_ASYNC_0() asm volatile("s_wait_asynccnt 0" ::: "memory")
#define WAIT_ASYNC_4() asm volatile("s_wait_asynccnt 4" ::: "memory")
#define WAIT_ASYNC_5() asm volatile("s_wait_asynccnt 5" ::: "memory")

// 16x32 WMMA A/B fragment per CDNA5 ISA 7.12.2 16-bit layout:
//   lanes 0-15 : row r=lane,    K = {0..7, 16..23}
//   lanes 16-31: row r=lane-16, K = {8..15, 24..31}
template <typename T>
__device__ __forceinline__ v16h load_frag(const T* base, long rstride, long kstride) {
  const int  lane = threadIdx.x & 31;
  const long r    = lane & 15;
  const long koff = (lane >> 4) * 8;
  v16h f;
#pragma unroll
  for (int j = 0; j < 8; ++j) {
    f[j]     = (_Float16)base[r * rstride + (koff + j) * kstride];
    f[j + 8] = (_Float16)base[r * rstride + (16 + koff + j) * kstride];
  }
  return f;
}

// ---------------------------------------------------------------------------
// 1) mask lengths: lens[0..3]=sum(x_mask[b]), lens[4..7]=sum(context_mask[b])
// ---------------------------------------------------------------------------
__global__ void lens_kernel(const float* __restrict__ xm,
                            const float* __restrict__ cm,
                            float* __restrict__ lens) {
  __shared__ float red[256];
  const int b = blockIdx.x, which = blockIdx.y;
  const float* p = which ? cm : xm;
  const int n = which ? LSEQ : TSEQ;
  float s = 0.f;
  for (int i = threadIdx.x; i < n; i += 256) s += p[b * n + i];
  red[threadIdx.x] = s;
  __syncthreads();
  for (int off = 128; off; off >>= 1) {
    if ((int)threadIdx.x < off) red[threadIdx.x] += red[threadIdx.x + off];
    __syncthreads();
  }
  if (threadIdx.x == 0) lens[which * BATCH + b] = red[0];
}

// ---------------------------------------------------------------------------
// 2a) transpose x (B,512,T) f32 -> Xh (B,T,512) f16, 32x32 LDS tiles
// ---------------------------------------------------------------------------
__global__ __launch_bounds__(256) void transpose_x_kernel(
    const float* __restrict__ x, _Float16* __restrict__ Xh) {
  __shared__ float tile[32][33];
  const int b  = blockIdx.z;
  const int t0 = blockIdx.x * 32;
  const int d0 = blockIdx.y * 32;
  const int r = threadIdx.x >> 3;          // 0..31
  const int c = (threadIdx.x & 7) * 4;     // 0,4,..,28
  const float4 v = *(const float4*)(x + ((size_t)b * D_MODEL + d0 + r) * TSEQ + t0 + c);
  tile[r][c + 0] = v.x; tile[r][c + 1] = v.y;
  tile[r][c + 2] = v.z; tile[r][c + 3] = v.w;
  __syncthreads();
  v4h o;
  o[0] = (_Float16)tile[c + 0][r];
  o[1] = (_Float16)tile[c + 1][r];
  o[2] = (_Float16)tile[c + 2][r];
  o[3] = (_Float16)tile[c + 3][r];
  *(v4h*)(Xh + ((size_t)b * TSEQ + t0 + r) * D_MODEL + d0 + c) = o;
}

// ---------------------------------------------------------------------------
// 2b) f32 -> f16 straight convert (context + the four weight matrices)
// ---------------------------------------------------------------------------
__global__ __launch_bounds__(256) void cvt_f16_kernel(
    const float* __restrict__ src, _Float16* __restrict__ dst, size_t n4) {
  const size_t i = ((size_t)blockIdx.x * 256 + threadIdx.x);
  if (i >= n4) return;
  const float4 v = ((const float4*)src)[i];
  v4h o; o[0] = (_Float16)v.x; o[1] = (_Float16)v.y;
  o[2] = (_Float16)v.z; o[3] = (_Float16)v.w;
  ((v4h*)dst)[i] = o;
}

// ---------------------------------------------------------------------------
// GEMM main loop: acc += A(16x512) * W(64x512)^T, this wave's 16-col slice.
// All-f16 LDS, K chunks of 64, double buffered, fully unrolled.
// Per-wave ASYNCcnt per chunk: 1 (A) + 4 (W) = 5.
// ---------------------------------------------------------------------------
__device__ __forceinline__ v8f gemm16_loop(
    _Float16 (&As)[2][16][64], _Float16 (&Ws)[2][64][64],
    const _Float16* __restrict__ Ag,   // [16][512] row-major tile base
    const _Float16* __restrict__ Wg) { // Wh + nt_block*512 (64 rows)
  const int tid = threadIdx.x;
  const int w   = tid >> 5;
  const int ar  = tid >> 3, aseg = (tid & 7) * 8;   // A: row, 8-half segment
  const int wr  = tid >> 1, wh   = (tid & 1) * 32;  // W: row, 32-half half

  v8f acc = {0.f, 0.f, 0.f, 0.f, 0.f, 0.f, 0.f, 0.f};

  auto issue = [&](int buf, int kc) {
    async_b128(lds_off(&As[buf][ar][aseg]), Ag + (size_t)ar * D_MODEL + kc + aseg);
#pragma unroll
    for (int j = 0; j < 4; ++j)
      async_b128(lds_off(&Ws[buf][wr][wh + j * 8]),
                 Wg + (size_t)wr * D_MODEL + kc + wh + j * 8);
  };

  issue(0, 0);
#pragma unroll
  for (int kc = 0; kc < D_MODEL; kc += 64) {
    const int buf = (kc >> 6) & 1;
    if (kc + 64 < D_MODEL) { issue(buf ^ 1, kc + 64); WAIT_ASYNC_5(); }
    else                   { WAIT_ASYNC_0(); }
    __syncthreads();
#pragma unroll
    for (int kk = 0; kk < 64; kk += 32) {
      v16h af = load_frag(&As[buf][0][kk], 64L, 1L);
      v16h bf = load_frag(&Ws[buf][w * 16][kk], 64L, 1L);
      acc = wmma16(af, bf, acc);
    }
    __syncthreads();  // buf fully consumed before it is refilled
  }
  return acc;
}

// ---------------------------------------------------------------------------
// 3) projection GEMM: Y = A . W^T + bias (f16 out).
//    VT=false: Y row-major (b, m, n).  VT=true (V proj): store transposed
//    per head: Vt[b][h][dd][l] so flash can consume d-major tiles directly.
// ---------------------------------------------------------------------------
template <bool VT>
__global__ __launch_bounds__(128) void gemm_proj_kernel(
    const _Float16* __restrict__ A, const _Float16* __restrict__ W,
    const float* __restrict__ bias, _Float16* __restrict__ Y) {
  __shared__ __align__(16) _Float16 As[2][16][64];
  __shared__ __align__(16) _Float16 Ws[2][64][64];
  const int b  = blockIdx.z;
  const int mt = blockIdx.x * 16;
  const int nb = blockIdx.y * 64;
  const int nt = nb + (threadIdx.x >> 5) * 16;

  v8f acc = gemm16_loop(As, Ws, A + ((size_t)b * TSEQ + mt) * D_MODEL,
                        W + (size_t)nb * D_MODEL);

  const int lane = threadIdx.x & 31;
  const int cc   = lane & 15;
  const int rb   = (lane >> 4) * 8;
  const int n    = nt + cc;
  const float bv = bias[n];
  if (VT) {
    const int h = n >> 6, dd = n & 63;
    _Float16* dst = Y + (((size_t)b * NHEADS + h) * HDIM + dd) * LSEQ + mt + rb;
#pragma unroll
    for (int i = 0; i < 8; ++i) dst[i] = (_Float16)(acc[i] + bv);  // contiguous in l
  } else {
#pragma unroll
    for (int i = 0; i < 8; ++i) {
      const size_t m = (size_t)(mt + rb + i);
      Y[((size_t)b * TSEQ + m) * D_MODEL + n] = (_Float16)(acc[i] + bv);
    }
  }
}

// ---------------------------------------------------------------------------
// 4) RoPE in-place on f16 Q or K: pairs (i, i+32) per 64-wide head
// ---------------------------------------------------------------------------
__global__ void rope_kernel(_Float16* __restrict__ Yh,
                            const float* __restrict__ lens, int seqlen) {
  const int t = blockIdx.x, b = blockIdx.y;
  const int h = threadIdx.x >> 5, i = threadIdx.x & 31;
  const float len = lens[b];
  const float inv = __powf(10000.0f, -(2.0f * (float)i) / 64.0f);
  const float fr  = ((float)t / len) * (10.0f * inv);
  const float c = __cosf(fr), s = __sinf(fr);
  const size_t base = ((size_t)b * seqlen + t) * D_MODEL + h * HDIM;
  const float x1 = (float)Yh[base + i];
  const float x2 = (float)Yh[base + 32 + i];
  Yh[base + i]      = (_Float16)(x1 * c - x2 * s);
  Yh[base + 32 + i] = (_Float16)(x1 * s + x2 * c);
}

// ---------------------------------------------------------------------------
// 5) flash attention: grid (T/64, H, B), block 128 (4 waves x 16 queries).
//    32-key K tiles (l-major) + V tiles (d-major) async-staged, double
//    buffered; 4 async instr/wave per chunk; last iteration peeled.
// ---------------------------------------------------------------------------
__global__ __launch_bounds__(128) void flash_kernel(
    const _Float16* __restrict__ Qh, const _Float16* __restrict__ Kh,
    const _Float16* __restrict__ Vt, const float* __restrict__ cmask,
    _Float16* __restrict__ Oh) {
  __shared__ __align__(16) _Float16 Kc[2][32][HDIM];   // [l][d]
  __shared__ __align__(16) _Float16 Vc[2][HDIM][32];   // [d][l]
  __shared__ __align__(16) _Float16 Pbuf[4][16][32];
  __shared__ float cmv[2][32];

  const int tid = threadIdx.x;
  const int b = blockIdx.z, h = blockIdx.y;
  const int w = tid >> 5, lane = tid & 31;
  const int qbase = blockIdx.x * 64 + w * 16;
  const float scale = 0.044194173824159216f;  // 1/sqrt(512)

  const size_t qoff = ((size_t)b * TSEQ + qbase) * D_MODEL + (size_t)h * HDIM;
  const v16h qf0 = load_frag(Qh + qoff,      (long)D_MODEL, 1L);
  const v16h qf1 = load_frag(Qh + qoff + 32, (long)D_MODEL, 1L);

  v8f o0 = {0.f,0.f,0.f,0.f,0.f,0.f,0.f,0.f}, o1 = o0, o2 = o0, o3 = o0;
  float mrow[8], lrow[8];
#pragma unroll
  for (int i = 0; i < 8; ++i) { mrow[i] = -1e30f; lrow[i] = 0.f; }

  const size_t krow0 = ((size_t)b * LSEQ) * D_MODEL + (size_t)h * HDIM;
  const size_t vrow0 = ((size_t)b * NHEADS + h) * HDIM * LSEQ;

  auto issue_kv = [&](int buf, int lb) {
#pragma unroll
    for (int part = 0; part < 2; ++part) {
      const int f = part * 1024 + tid * 8;  // flat half index in 2048-half tile
      // K tile: [l=f>>6][d=f&63]
      async_b128(lds_off(&Kc[buf][0][0]) + (unsigned)f * 2,
                 Kh + krow0 + (size_t)(lb + (f >> 6)) * D_MODEL + (f & 63));
      // V tile: [d=f>>5][l=f&31] from d-major Vt
      async_b128(lds_off(&Vc[buf][0][0]) + (unsigned)f * 2,
                 Vt + vrow0 + (size_t)(f >> 5) * LSEQ + lb + (f & 31));
    }
    if (tid < 32) cmv[buf][tid] = cmask[(size_t)b * LSEQ + lb + tid];
  };

  auto compute = [&](int buf) {
    // ---- S = Q K^T for two 16-key sub-tiles ----
    v8f s0 = {0.f,0.f,0.f,0.f,0.f,0.f,0.f,0.f}, s1 = s0;
    s0 = wmma16(qf0, load_frag(&Kc[buf][0][0],   (long)HDIM, 1L), s0);
    s0 = wmma16(qf1, load_frag(&Kc[buf][0][32],  (long)HDIM, 1L), s0);
    s1 = wmma16(qf0, load_frag(&Kc[buf][16][0],  (long)HDIM, 1L), s1);
    s1 = wmma16(qf1, load_frag(&Kc[buf][16][32], (long)HDIM, 1L), s1);

    const float cm0 = cmv[buf][lane & 15];
    const float cm1 = cmv[buf][16 + (lane & 15)];

    // ---- online softmax (rows live per (vgpr idx, lane-half)) ----
    float p0f[8], p1f[8], rscale[8];
#pragma unroll
    for (int i = 0; i < 8; ++i) {
      float a0 = s0[i] * scale; if (cm0 == 0.f) a0 = -1e30f;
      float a1 = s1[i] * scale; if (cm1 == 0.f) a1 = -1e30f;
      float mx = fmaxf(a0, a1);
      mx = fmaxf(mx, __shfl_xor(mx, 1));
      mx = fmaxf(mx, __shfl_xor(mx, 2));
      mx = fmaxf(mx, __shfl_xor(mx, 4));
      mx = fmaxf(mx, __shfl_xor(mx, 8));
      const float mnew = fmaxf(mrow[i], mx);
      const float sc2  = __expf(mrow[i] - mnew);
      mrow[i] = mnew;
      const float e0 = __expf(a0 - mnew);
      const float e1 = __expf(a1 - mnew);
      float rs = e0 + e1;
      rs += __shfl_xor(rs, 1);
      rs += __shfl_xor(rs, 2);
      rs += __shfl_xor(rs, 4);
      rs += __shfl_xor(rs, 8);
      lrow[i] = lrow[i] * sc2 + rs;
      rscale[i] = sc2;
      p0f[i] = e0; p1f[i] = e1;
    }
#pragma unroll
    for (int i = 0; i < 8; ++i) {
      o0[i] *= rscale[i]; o1[i] *= rscale[i];
      o2[i] *= rscale[i]; o3[i] *= rscale[i];
    }

    // ---- P: C layout -> A layout via per-wave LDS tile (same-wave order) ----
    const int rb = (lane >> 4) * 8, cc = lane & 15;
#pragma unroll
    for (int i = 0; i < 8; ++i) {
      Pbuf[w][rb + i][cc]      = (_Float16)p0f[i];
      Pbuf[w][rb + i][16 + cc] = (_Float16)p1f[i];
    }
    const v16h pa = load_frag(&Pbuf[w][0][0], 32L, 1L);

    // ---- O += P V : B frag [n][k] = Vc[nt*16+n][k] (contiguous k) ----
    o0 = wmma16(pa, load_frag(&Vc[buf][0][0],  32L, 1L), o0);
    o1 = wmma16(pa, load_frag(&Vc[buf][16][0], 32L, 1L), o1);
    o2 = wmma16(pa, load_frag(&Vc[buf][32][0], 32L, 1L), o2);
    o3 = wmma16(pa, load_frag(&Vc[buf][48][0], 32L, 1L), o3);
  };

  issue_kv(0, 0);
  int buf = 0;
  for (int lb = 0; lb < LSEQ - 32; lb += 32) {
    issue_kv(buf ^ 1, lb + 32);
    WAIT_ASYNC_4();
    __syncthreads();
    compute(buf);
    __syncthreads();  // buf fully consumed before it is refilled
    buf ^= 1;
  }
  WAIT_ASYNC_0();
  __syncthreads();
  compute(buf);

  const int rb = (lane >> 4) * 8, cc = lane & 15;
#pragma unroll
  for (int i = 0; i < 8; ++i) {
    const float inv_l = 1.0f / fmaxf(lrow[i], 1e-20f);
    const size_t t = (size_t)(qbase + rb + i);
    const size_t base = ((size_t)b * TSEQ + t) * D_MODEL + (size_t)h * HDIM;
    Oh[base +  0 + cc] = (_Float16)(o0[i] * inv_l);
    Oh[base + 16 + cc] = (_Float16)(o1[i] * inv_l);
    Oh[base + 32 + cc] = (_Float16)(o2[i] * inv_l);
    Oh[base + 48 + cc] = (_Float16)(o3[i] * inv_l);
  }
}

// ---------------------------------------------------------------------------
// 6) out = (O @ Wo^T + b) * x_mask, stored transposed as (B, 512, T) f32
// ---------------------------------------------------------------------------
__global__ __launch_bounds__(128) void out_proj_kernel(
    const _Float16* __restrict__ Oh, const _Float16* __restrict__ Wo,
    const float* __restrict__ Wo_b, const float* __restrict__ xmask,
    float* __restrict__ out) {
  __shared__ __align__(16) _Float16 As[2][16][64];
  __shared__ __align__(16) _Float16 Ws[2][64][64];
  const int b  = blockIdx.z;
  const int mt = blockIdx.x * 16;
  const int nb = blockIdx.y * 64;
  const int nt = nb + (threadIdx.x >> 5) * 16;

  v8f acc = gemm16_loop(As, Ws, Oh + ((size_t)b * TSEQ + mt) * D_MODEL,
                        Wo + (size_t)nb * D_MODEL);

  const int lane = threadIdx.x & 31;
  const int cc   = lane & 15;
  const int rb   = (lane >> 4) * 8;
  const float bv = Wo_b[nt + cc];
#pragma unroll
  for (int i = 0; i < 8; ++i) {
    const int t = mt + rb + i;
    const float v = (acc[i] + bv) * xmask[(size_t)b * TSEQ + t];
    out[((size_t)b * D_MODEL + nt + cc) * TSEQ + t] = v;
  }
}

// ---------------------------------------------------------------------------
extern "C" void kernel_launch(void* const* d_in, const int* in_sizes, int n_in,
                              void* d_out, int out_size, void* d_ws, size_t ws_size,
                              hipStream_t stream) {
  const float* x        = (const float*)d_in[0];   // (B, 512, T)
  const float* context  = (const float*)d_in[1];   // (B, L, 512)
  const float* x_mask   = (const float*)d_in[2];   // (B, 1, T)
  const float* ctx_mask = (const float*)d_in[3];   // (B, 1, L)
  const float* Wq_w = (const float*)d_in[4];
  const float* Wq_b = (const float*)d_in[5];
  const float* Wk_w = (const float*)d_in[6];
  const float* Wk_b = (const float*)d_in[7];
  const float* Wv_w = (const float*)d_in[8];
  const float* Wv_b = (const float*)d_in[9];
  const float* Wo_w = (const float*)d_in[10];
  const float* Wo_b = (const float*)d_in[11];
  float* out = (float*)d_out;

  // ws: lens | Xh(=Oh) | Ch | Qh | Kh | Vt (8 MiB each) | Wqh|Wkh|Wvh|Woh (512 KiB each)
  char* ws = (char*)d_ws;
  float* lens = (float*)ws;
  const size_t NB = (size_t)BATCH * TSEQ * D_MODEL;  // 4,194,304 elements
  const size_t NW = (size_t)D_MODEL * D_MODEL;       // 262,144 elements
  _Float16* Xh  = (_Float16*)(ws + 256);
  _Float16* Ch  = Xh + NB;
  _Float16* Qh  = Ch + NB;
  _Float16* Kh  = Qh + NB;
  _Float16* Vt  = Kh + NB;   // (B, H, 64, L) d-major
  _Float16* Wqh = Vt + NB;
  _Float16* Wkh = Wqh + NW;
  _Float16* Wvh = Wkh + NW;
  _Float16* Woh = Wvh + NW;
  _Float16* Oh  = Xh;  // Xh dead after Q projection; flash runs later

  lens_kernel<<<dim3(BATCH, 2), 256, 0, stream>>>(x_mask, ctx_mask, lens);

  transpose_x_kernel<<<dim3(TSEQ / 32, D_MODEL / 32, BATCH), 256, 0, stream>>>(x, Xh);
  cvt_f16_kernel<<<dim3((unsigned)(NB / 4 / 256)), 256, 0, stream>>>(context, Ch, NB / 4);
  cvt_f16_kernel<<<dim3((unsigned)(NW / 4 / 256)), 256, 0, stream>>>(Wq_w, Wqh, NW / 4);
  cvt_f16_kernel<<<dim3((unsigned)(NW / 4 / 256)), 256, 0, stream>>>(Wk_w, Wkh, NW / 4);
  cvt_f16_kernel<<<dim3((unsigned)(NW / 4 / 256)), 256, 0, stream>>>(Wv_w, Wvh, NW / 4);
  cvt_f16_kernel<<<dim3((unsigned)(NW / 4 / 256)), 256, 0, stream>>>(Wo_w, Woh, NW / 4);

  gemm_proj_kernel<false><<<dim3(TSEQ / 16, D_MODEL / 64, BATCH), 128, 0, stream>>>(
      Xh, Wqh, Wq_b, Qh);
  gemm_proj_kernel<false><<<dim3(LSEQ / 16, D_MODEL / 64, BATCH), 128, 0, stream>>>(
      Ch, Wkh, Wk_b, Kh);
  gemm_proj_kernel<true><<<dim3(LSEQ / 16, D_MODEL / 64, BATCH), 128, 0, stream>>>(
      Ch, Wvh, Wv_b, Vt);

  rope_kernel<<<dim3(TSEQ, BATCH), 256, 0, stream>>>(Qh, lens + 0,     TSEQ);
  rope_kernel<<<dim3(LSEQ, BATCH), 256, 0, stream>>>(Kh, lens + BATCH, LSEQ);

  flash_kernel<<<dim3(TSEQ / 64, NHEADS, BATCH), 128, 0, stream>>>(
      Qh, Kh, Vt, ctx_mask, Oh);

  out_proj_kernel<<<dim3(TSEQ / 16, D_MODEL / 64, BATCH), 128, 0, stream>>>(
      Oh, Woh, Wo_b, x_mask, out);
}
